// QuantLinear_26293789786976
// MI455X (gfx1250) — matrix operand order repaired and checked by reading, not compiled, not run on hardware
//
#include <hip/hip_runtime.h>

// ---- dtypes for WMMA fragments (wave32, gfx1250) ----
typedef __attribute__((ext_vector_type(16))) _Float16 v16h;
typedef __attribute__((ext_vector_type(4)))  _Float16 v4h;
typedef __attribute__((ext_vector_type(2)))  _Float16 v2h;
typedef __attribute__((ext_vector_type(8)))  float    v8f;
typedef __attribute__((ext_vector_type(4)))  float    v4f;

#define TOKENS     8192
#define IN_F       4096
#define OUT_F      4096
#define GROUPSIZE  128
#define N_GROUPS   (IN_F / GROUPSIZE)   // 32

#define BM 256
#define BN 128
// LDS A-tile stride in halves (128 + 8 pad keeps 16B alignment: 136*2=272=16*17)
#define LDS_STRIDE 136
#define TILE_H     (BM * LDS_STRIDE)    // 34,816 halves per buffer

// Unpack 8 nibbles of one qweight word into 4 packed-f16 pairs.
// Pair s holds (nib_s, nib_{s+4}) as f16 via the 1024-bias magic exponent:
//   ((w>>4s) & 0x000F000F) | 0x64006400  ==  f16x2(1024+nib_s, 1024+nib_{s+4})
// Then (p - zpk) * spk = (nib - z) * scale, with (nib - z) exact in f16.
__device__ __forceinline__ void dequant8(unsigned w, v2h zpk, v2h spk, v2h* dst) {
#pragma unroll
    for (int s = 0; s < 4; ++s) {
        unsigned u = ((w >> (4 * s)) & 0x000F000Fu) | 0x64006400u;
        v2h p = __builtin_bit_cast(v2h, u);
        dst[s] = (p - zpk) * spk;
    }
}

// Stage x[m0:m0+256, k0:k0+128] (f32) into LDS as f16, with the matching
// within-8 K interleave [k0,k4,k1,k5,k2,k6,k3,k7] so A fragments stay
// contiguous ds_load_b128 while B uses the free nibble permutation.
__device__ __forceinline__ void stage_tile(const float* __restrict__ x,
                                           _Float16* __restrict__ sbuf,
                                           int m0, int k0, int tid) {
#pragma unroll
    for (int i = 0; i < 16; ++i) {
        const int gidx = tid + i * 256;        // 4096 8-wide groups per tile
        const int row  = gidx >> 4;            // 16 groups per row
        const int qo   = (gidx & 15) << 3;     // logical k offset in tile
        const float* xp = x + (size_t)(m0 + row) * IN_F + k0 + qo;
        const v4f f0 = *(const v4f*)xp;        // k = qo+0..3
        const v4f f1 = *(const v4f*)(xp + 4);  // k = qo+4..7
        v4h o0, o1;
        o0[0] = (_Float16)f0[0]; o0[1] = (_Float16)f1[0];
        o0[2] = (_Float16)f0[1]; o0[3] = (_Float16)f1[1];
        o1[0] = (_Float16)f0[2]; o1[1] = (_Float16)f1[2];
        o1[2] = (_Float16)f0[3]; o1[3] = (_Float16)f1[3];
        _Float16* sp = &sbuf[row * LDS_STRIDE + qo];
        *(v4h*)(sp)     = o0;
        *(v4h*)(sp + 4) = o1;
    }
}

__global__ __launch_bounds__(256)
void gptq_wmma_f16(const float*    __restrict__ x,        // [8192,4096] f32
                   const int*      __restrict__ qweight,  // [512,4096] i32 (8 nibbles along K)
                   const unsigned* __restrict__ qzeros,   // [32,512]  u32 (8 nibbles along N)
                   const float*    __restrict__ scales,   // [32,4096] f32
                   const float*    __restrict__ bias,     // [4096]    f32
                   float*          __restrict__ out)      // [8192,4096] f32
{
    __shared__ _Float16 sA[2 * TILE_H];   // 139,264 B double-buffered A tile

    const int tid  = threadIdx.x;
    const int lane = tid & 31;
    const int wave = tid >> 5;
    const int wm   = wave >> 2;        // 0..1  (128 rows each)
    const int wn   = wave & 3;         // 0..3  (32 cols each)
    const int hs   = lane >> 4;        // lane-half select
    const int ml   = lane & 15;

    const int m0 = blockIdx.y * BM;
    const int n0 = blockIdx.x * BN;

    int ncol[2];
    ncol[0] = n0 + wn * 32 + ml;
    ncol[1] = ncol[0] + 16;

    v8f acc[8][2];
#pragma unroll
    for (int tm = 0; tm < 8; ++tm)
#pragma unroll
        for (int t = 0; t < 2; ++t)
            acc[tm][t] = (v8f){0.f, 0.f, 0.f, 0.f, 0.f, 0.f, 0.f, 0.f};

    stage_tile(x, sA, m0, 0, tid);     // prologue: tile for group 0

    for (int g = 0; g < N_GROUPS; ++g) {
        __syncthreads();               // staging of buf[g&1] complete
        const _Float16* cur = sA + (size_t)(g & 1) * TILE_H;

        // ---- preload ALL qweight words for this group (one load clause) ----
        // per kb, this lane-half needs rows g*16 + kb*4 + hs*2 + {0,1}
        unsigned wq[2][8];
#pragma unroll
        for (int t = 0; t < 2; ++t)
#pragma unroll
            for (int kb = 0; kb < 4; ++kb) {
                const size_t r = (size_t)(g * 16 + kb * 4 + hs * 2) * OUT_F + ncol[t];
                wq[t][kb * 2]     = (unsigned)qweight[r];
                wq[t][kb * 2 + 1] = (unsigned)qweight[r + OUT_F];
            }

        // overlap: stage next group's x tile into the other buffer
        if (g + 1 < N_GROUPS)
            stage_tile(x, sA + (size_t)((g + 1) & 1) * TILE_H, m0, (g + 1) * GROUPSIZE, tid);

        // ---- per-group packed zero-point & scale for this lane's columns ----
        v2h zpk[2], spk[2];
#pragma unroll
        for (int t = 0; t < 2; ++t) {
            const int n = ncol[t];
            const unsigned zw = qzeros[g * (OUT_F / 8) + (n >> 3)];
            const unsigned z  = ((zw >> ((n & 7) * 4)) & 15u) + 1u;   // 1..16
            zpk[t] = __builtin_bit_cast(v2h, 0x64006400u | z | (z << 16));
            const _Float16 sh = (_Float16)scales[g * OUT_F + n];
            spk[t][0] = sh; spk[t][1] = sh;
        }

        // ---- 4 K-blocks of 32 inside this group ----
#pragma unroll
        for (int kb = 0; kb < 4; ++kb) {
            // B fragments from preloaded words
            union { v16h v; v2h p[8]; } bf[2];
#pragma unroll
            for (int t = 0; t < 2; ++t) {
                dequant8(wq[t][kb * 2],     zpk[t], spk[t], &bf[t].p[0]);
                dequant8(wq[t][kb * 2 + 1], zpk[t], spk[t], &bf[t].p[4]);
            }

            const int ks = kb * 32 + hs * 16;
            // Two tm-halves of 4: 8 ds_load_b128 issued, then 8 WMMAs.
            // Next half's loads issue while this half's WMMAs drain the
            // matrix pipe, so LDS latency stays hidden with only a
            // 4-fragment (32 VGPR) A buffer — keeps total < 256 VGPRs.
#pragma unroll
            for (int half = 0; half < 2; ++half) {
                union { v16h v; v4h q[4]; } af[4];
#pragma unroll
                for (int i = 0; i < 4; ++i) {
                    const int row = wm * 128 + (half * 4 + i) * 16 + ml;
                    const _Float16* ap = &cur[row * LDS_STRIDE + ks];
                    af[i].q[0] = *(const v4h*)(ap);
                    af[i].q[1] = *(const v4h*)(ap + 4);
                    af[i].q[2] = *(const v4h*)(ap + 8);
                    af[i].q[3] = *(const v4h*)(ap + 12);
                }
#pragma unroll
                for (int i = 0; i < 4; ++i)
#pragma unroll
                    for (int t = 0; t < 2; ++t)
                        acc[half * 4 + i][t] = __builtin_amdgcn_wmma_f32_16x16x32_f16(
                            /*neg_a=*/false, af[i].v, /*neg_b=*/false, bf[t].v,
                            /*c_mod=*/(short)0, acc[half * 4 + i][t],
                            /*reuse_a=*/false, /*reuse_b=*/false);
            }
        }
    }

    // ---- epilogue: D layout -> VGPR j holds M = j + 8*hs, N = ml ----
#pragma unroll
    for (int t = 0; t < 2; ++t) {
        const float bv = bias[ncol[t]];
#pragma unroll
        for (int tm = 0; tm < 8; ++tm) {
            const int rowbase = m0 + wm * 128 + tm * 16 + hs * 8;
#pragma unroll
            for (int j = 0; j < 8; ++j)
                out[(size_t)(rowbase + j) * OUT_F + ncol[t]] = acc[tm][t][j] + bv;
        }
    }
}

extern "C" void kernel_launch(void* const* d_in, const int* in_sizes, int n_in,
                              void* d_out, int out_size, void* d_ws, size_t ws_size,
                              hipStream_t stream) {
    const float*    x       = (const float*)d_in[0];
    const int*      qweight = (const int*)d_in[1];
    const unsigned* qzeros  = (const unsigned*)d_in[2];
    const float*    scales  = (const float*)d_in[3];
    // d_in[4] = g_idx (int32) — trivially k // 128, not needed
    const float*    bias    = (const float*)d_in[5];
    float*          out     = (float*)d_out;

    dim3 grid(OUT_F / BN, TOKENS / BM);   // (32, 32)
    dim3 block(256);
    gptq_wmma_f16<<<grid, block, 0, stream>>>(x, qweight, qzeros, scales, bias, out);
}
